// GATNet_12996571038301
// MI455X (gfx1250) — matrix operand order repaired and checked by reading, not compiled, not run on hardware
//
#include <hip/hip_runtime.h>
#include <hip/hip_bf16.h>
#include <math.h>

typedef __attribute__((ext_vector_type(16))) _Float16 v16h;
typedef __attribute__((ext_vector_type(8)))  float    v8f;

// ---------------------------------------------------------------------------
// WMMA GEMM, software-pipelined. Static requirements (met by layers 2-6 + fc1):
//   K % 64 == 0, Nc % 16 == 0, M % 16 == 0, lda == 3072.
// One wave computes a 64x16 strip: 4 M-tiles share one B fragment per K-step.
// Two named staging buffers double-buffer the loads so each compute stage
// waits only while the *other* stage's loads are in flight.
// NC is a template constant -> B loads / C stores use immediate offsets.
// __launch_bounds__(32, 1): single-wave block, full VGPR budget -> no spills.
// Fragment layouts per CDNA5 ISA 7.12.2 (wave32, 16-bit A 16x32 / B 32x16).
// ---------------------------------------------------------------------------
template <int NC>
__device__ __forceinline__ void load_b(const float* __restrict__ bp, int kofs, float (&br)[16])
{
#pragma unroll
    for (int j = 0; j < 16; ++j) br[j] = bp[(long long)(kofs + j + (j >= 8 ? 8 : 0)) * NC];
}

__device__ __forceinline__ void load_a(const float* const (&ap)[4], int kofs, float4 (&ar)[4][4])
{
#pragma unroll
    for (int t = 0; t < 4; ++t) {
        ar[t][0] = *(const float4*)(ap[t] + kofs);
        ar[t][1] = *(const float4*)(ap[t] + kofs + 4);
        ar[t][2] = *(const float4*)(ap[t] + kofs + 16);
        ar[t][3] = *(const float4*)(ap[t] + kofs + 20);
    }
}

__device__ __forceinline__ void compute4(const float (&br)[16], const float4 (&ar)[4][4],
                                         v8f (&acc)[4])
{
    v16h bf;
#pragma unroll
    for (int j = 0; j < 16; ++j) bf[j] = (_Float16)br[j];
#pragma unroll
    for (int t = 0; t < 4; ++t) {
        v16h af;
#pragma unroll
        for (int q = 0; q < 4; ++q) {
            af[4 * q + 0] = (_Float16)ar[t][q].x;
            af[4 * q + 1] = (_Float16)ar[t][q].y;
            af[4 * q + 2] = (_Float16)ar[t][q].z;
            af[4 * q + 3] = (_Float16)ar[t][q].w;
        }
        acc[t] = __builtin_amdgcn_wmma_f32_16x16x32_f16(
            false, af, false, bf, (short)0, acc[t], false, false);
    }
}

template <int NC>
__global__ void __launch_bounds__(32, 1)
wmma_gemm4t(const float* __restrict__ A, const float* __restrict__ B,
            float* __restrict__ C, int M, int K, int aoff)
{
    constexpr int LDA = 3072;
    const int lane = threadIdx.x;              // 0..31
    const int n0   = blockIdx.x * 16;
    const int m0   = blockIdx.y * 64;

    const int  lm  = lane & 15;
    const bool hi  = (lane & 16) != 0;
    const int  bn  = n0 + lm;                  // B/D column (NC % 16 == 0)
    const int  akb = hi ? 8 : 0;               // A: hi lanes hold K+8..15 / K+24..31
    const int  bkb = hi ? 8 : 0;               // B: hi lanes hold K+8..15 / K+24..31

    // Per-tile A pointers, rows clamped into [0, M) so loads never go OOB.
    const float* ap[4];
#pragma unroll
    for (int t = 0; t < 4; ++t) {
        int am = m0 + t * 16 + lm;
        am = (am < M) ? am : (M - 1);
        ap[t] = A + (long long)am * LDA + aoff + akb;
    }
    const float* bp = B + (long long)bkb * NC + bn;

    v8f acc[4] = {};

    // Two pipeline stages, statically named so staging stays in VGPRs.
    float  b0r[16], b1r[16];
    float4 a0r[4][4], a1r[4][4];

    load_b<NC>(bp, 0, b0r);
    load_a(ap, 0, a0r);

    for (int k0 = 0; k0 < K; k0 += 64) {
        // stage 1 loads (this 64-block, K+32..63)
        load_b<NC>(bp, 32, b1r);
        load_a(ap, 32, a1r);

        // compute stage 0 (waits only while stage-1 loads are in flight)
        compute4(b0r, a0r, acc);

        // advance to next 64-block, then issue its stage-0 loads
        bp += (long long)64 * NC;
#pragma unroll
        for (int t = 0; t < 4; ++t) ap[t] += 64;
        if (k0 + 64 < K) {
            load_b<NC>(bp, 0, b0r);
            load_a(ap, 0, a0r);
        }

        // compute stage 1 (next stage-0 loads still in flight)
        compute4(b1r, a1r, acc);
    }

    // D layout: lane -> N = bn ; acc[t][r] -> M = m0 + t*16 + r + 8*hi.
    // M % 16 == 0 -> tile validity is uniform; plain scalar branch.
#pragma unroll
    for (int t = 0; t < 4; ++t) {
        if (m0 + t * 16 < M) {
            const int mb = m0 + t * 16 + (hi ? 8 : 0);
            float* cp = C + (long long)mb * NC + bn;
#pragma unroll
            for (int r = 0; r < 8; ++r) cp[(long long)r * NC] = acc[t][r];
        }
    }
}

// Layer 1 transform: K = 3 (no WMMA shape fits; 0.01% of FLOPs).
__global__ void lin_k3(const float* __restrict__ x, const float* __restrict__ W,
                       float* __restrict__ h, int N, int width)
{
    long long i = (long long)blockIdx.x * blockDim.x + threadIdx.x;
    if (i >= (long long)N * width) return;
    int n = (int)(i / width), j = (int)(i % width);
    float acc = x[n * 3 + 0] * W[j] + x[n * 3 + 1] * W[width + j] + x[n * 3 + 2] * W[2 * width + j];
    h[i] = acc;
}

// ---------------------------------------------------------------------------
// GAT attention helpers
// ---------------------------------------------------------------------------
__device__ __forceinline__ float leaky02(float v) { return v > 0.0f ? v : 0.2f * v; }

__device__ __forceinline__ void atomicMaxF(float* addr, float val)
{
    unsigned int* ai = (unsigned int*)addr;
    unsigned int old = *ai;
    while (__uint_as_float(old) < val) {
        unsigned int assumed = old;
        old = atomicCAS(ai, assumed, __float_as_uint(val));
        if (old == assumed) break;
    }
}

__global__ void attn_logits(const float* __restrict__ h,
                            const float* __restrict__ a_s, const float* __restrict__ a_d,
                            float* __restrict__ als, float* __restrict__ ald,
                            int N, int H, int F)
{
    int idx = blockIdx.x * blockDim.x + threadIdx.x;
    if (idx >= N * H) return;
    int n = idx / H, hd = idx % H;
    const float* hp  = h + (long long)n * H * F + (long long)hd * F;
    const float* asp = a_s + hd * F;
    const float* adp = a_d + hd * F;
    float ss = 0.0f, sd = 0.0f;
    for (int f = 0; f < F; ++f) { float v = hp[f]; ss += v * asp[f]; sd += v * adp[f]; }
    als[idx] = ss; ald[idx] = sd;
}

__global__ void init_attn_state(float* __restrict__ m, float* __restrict__ denom, int count)
{
    int i = blockIdx.x * blockDim.x + threadIdx.x;
    if (i < count) { m[i] = -1e30f; denom[i] = 0.0f; }
}

__global__ void init_out_bias(float* __restrict__ z, const float* __restrict__ b,
                              int N, int width, int ldz, int coff)
{
    long long i = (long long)blockIdx.x * blockDim.x + threadIdx.x;
    if (i >= (long long)N * width) return;
    int n = (int)(i / width), j = (int)(i % width);
    z[(long long)n * ldz + coff + j] = b[j];
}

__global__ void edge_max(const int* __restrict__ src, const int* __restrict__ dst,
                         int E, int EDGES,
                         const float* __restrict__ als, const float* __restrict__ ald,
                         float* __restrict__ m, int H)
{
    long long idx = (long long)blockIdx.x * blockDim.x + threadIdx.x;
    if (idx >= (long long)EDGES * H) return;
    int e = (int)(idx / H), hd = (int)(idx % H);
    int s = (e < E) ? src[e] : (e - E);
    int d = (e < E) ? dst[e] : (e - E);
    float ev = leaky02(als[s * H + hd] + ald[d * H + hd]);
    atomicMaxF(&m[d * H + hd], ev);
}

__global__ void edge_exp(const int* __restrict__ src, const int* __restrict__ dst,
                         int E, int EDGES,
                         const float* __restrict__ als, const float* __restrict__ ald,
                         const float* __restrict__ m, float* __restrict__ exbuf,
                         float* __restrict__ denom, int H)
{
    long long idx = (long long)blockIdx.x * blockDim.x + threadIdx.x;
    if (idx >= (long long)EDGES * H) return;
    int e = (int)(idx / H), hd = (int)(idx % H);
    int s = (e < E) ? src[e] : (e - E);
    int d = (e < E) ? dst[e] : (e - E);
    float ev = leaky02(als[s * H + hd] + ald[d * H + hd]);
    float ex = expf(ev - m[d * H + hd]);
    exbuf[idx] = ex;
    atomicAdd(&denom[d * H + hd], ex);
}

__global__ void edge_aggr(const int* __restrict__ src, const int* __restrict__ dst,
                          int E, int EDGES,
                          const float* __restrict__ h, const float* __restrict__ exbuf,
                          const float* __restrict__ denom, float* __restrict__ z,
                          int H, int F, int ldz, int coff)
{
    const int width = H * F;
    long long idx = (long long)blockIdx.x * blockDim.x + threadIdx.x;
    if (idx >= (long long)EDGES * width) return;
    int e = (int)(idx / width), j = (int)(idx % width);
    int hd = j / F;
    int s = (e < E) ? src[e] : (e - E);
    int d = (e < E) ? dst[e] : (e - E);
    float alpha = exbuf[(long long)e * H + hd] / (denom[d * H + hd] + 1e-16f);
    atomicAdd(&z[(long long)d * ldz + coff + j], h[(long long)s * width + j] * alpha);
}

// ---------------------------------------------------------------------------
// MLP head
// ---------------------------------------------------------------------------
__global__ void bias_bn_relu(float* __restrict__ z1, const float* __restrict__ fb,
                             const float* __restrict__ g, const float* __restrict__ bb,
                             const float* __restrict__ mm, const float* __restrict__ vv,
                             int N, int C)
{
    long long i = (long long)blockIdx.x * blockDim.x + threadIdx.x;
    if (i >= (long long)N * C) return;
    int j = (int)(i % C);
    float v = z1[i] + fb[j];
    v = (v - mm[j]) * rsqrtf(vv[j] + 1e-5f) * g[j] + bb[j];
    z1[i] = fmaxf(v, 0.0f);
}

__global__ void pool_zero(float* __restrict__ sums, float* __restrict__ cnt, int G, int C)
{
    int i = blockIdx.x * blockDim.x + threadIdx.x;
    if (i < G * C) sums[i] = 0.0f;
    if (i < G) cnt[i] = 0.0f;
}

__global__ void pool_cnt(const int* __restrict__ batch, float* __restrict__ cnt, int N)
{
    int n = blockIdx.x * blockDim.x + threadIdx.x;
    if (n < N) atomicAdd(&cnt[batch[n]], 1.0f);
}

__global__ void pool_sum(const int* __restrict__ batch, const float* __restrict__ z1,
                         float* __restrict__ sums, int N, int C)
{
    long long i = (long long)blockIdx.x * blockDim.x + threadIdx.x;
    if (i >= (long long)N * C) return;
    int n = (int)(i / C), j = (int)(i % C);
    atomicAdd(&sums[batch[n] * C + j], z1[(long long)n * C + j]);
}

__global__ void pool_div(const float* __restrict__ sums, const float* __restrict__ cnt,
                         float* __restrict__ zp, int G, int C)
{
    int i = blockIdx.x * blockDim.x + threadIdx.x;
    if (i >= G * C) return;
    zp[i] = sums[i] / fmaxf(cnt[i / C], 1.0f);
}

__global__ void fc2_relu(const float* __restrict__ zp, const float* __restrict__ Wm,
                         const float* __restrict__ b, float* __restrict__ z2,
                         int G, int Cin, int Cout)
{
    int i = blockIdx.x * blockDim.x + threadIdx.x;
    if (i >= G * Cout) return;
    int g = i / Cout, k = i % Cout;
    float acc = b[k];
    for (int j = 0; j < Cin; ++j) acc += zp[g * Cin + j] * Wm[j * Cout + k];
    z2[i] = fmaxf(acc, 0.0f);
}

__global__ void lin_sigmoid(const float* __restrict__ z2, const float* __restrict__ Wm,
                            const float* __restrict__ b, float* __restrict__ out,
                            int G, int Cin)
{
    int g = blockIdx.x * blockDim.x + threadIdx.x;
    if (g >= G) return;
    float acc = b[0];
    for (int j = 0; j < Cin; ++j) acc += z2[g * Cin + j] * Wm[j];
    out[g] = 1.0f / (1.0f + expf(-acc));
}

// ---------------------------------------------------------------------------
// Host orchestration
// ---------------------------------------------------------------------------
static inline unsigned gsz(long long n, int b) { return (unsigned)((n + b - 1) / b); }

static void launch_gemm(int NC, const float* A, const float* B, float* C,
                        int M, int K, int aoff, hipStream_t stream)
{
    dim3 gg(NC / 16, (M + 63) / 64);
    dim3 bb(32);
    switch (NC) {
    case 128:  wmma_gemm4t<128><<<gg, bb, 0, stream>>>(A, B, C, M, K, aoff);  break;
    case 256:  wmma_gemm4t<256><<<gg, bb, 0, stream>>>(A, B, C, M, K, aoff);  break;
    case 384:  wmma_gemm4t<384><<<gg, bb, 0, stream>>>(A, B, C, M, K, aoff);  break;
    case 512:  wmma_gemm4t<512><<<gg, bb, 0, stream>>>(A, B, C, M, K, aoff);  break;
    case 1024: wmma_gemm4t<1024><<<gg, bb, 0, stream>>>(A, B, C, M, K, aoff); break;
    default:   break;  // not used
    }
}

extern "C" void kernel_launch(void* const* d_in, const int* in_sizes, int n_in,
                              void* d_out, int out_size, void* d_ws, size_t ws_size,
                              hipStream_t stream)
{
    (void)n_in; (void)out_size; (void)ws_size;

    const float* x     = (const float*)d_in[0];
    const int*   src   = (const int*)d_in[1];
    const int*   dst   = (const int*)d_in[2];
    const int*   batch = (const int*)d_in[3];
    const float *Wl[6], *ASl[6], *ADl[6], *Bl[6];
    for (int i = 0; i < 6; ++i) {
        Wl[i]  = (const float*)d_in[4 + 4 * i];
        ASl[i] = (const float*)d_in[5 + 4 * i];
        ADl[i] = (const float*)d_in[6 + 4 * i];
        Bl[i]  = (const float*)d_in[7 + 4 * i];
    }
    const float* fc1_W = (const float*)d_in[28];
    const float* fc1_b = (const float*)d_in[29];
    const float* bn_g  = (const float*)d_in[30];
    const float* bn_b  = (const float*)d_in[31];
    const float* bn_m  = (const float*)d_in[32];
    const float* bn_v  = (const float*)d_in[33];
    const float* fc2_W = (const float*)d_in[34];
    const float* fc2_b = (const float*)d_in[35];
    const float* lin_W = (const float*)d_in[36];
    const float* lin_b = (const float*)d_in[37];

    const int N = in_sizes[0] / 3;
    const int E = in_sizes[1];
    const int G = 64;
    const int EDGES = E + N;
    const int ZC = 3072;

    static const int HH[6] = {8, 8, 8, 16, 16, 16};
    static const int CO[6] = {16, 16, 32, 32, 64, 64};
    const int HMAX = 16, WMAX = 1024;

    // ---- carve workspace ----
    char* wsp = (char*)d_ws;
    auto carve = [&](size_t bytes) -> char* {
        char* p = wsp;
        wsp += (bytes + 255) & ~(size_t)255;
        return p;
    };
    float* z     = (float*)carve((size_t)N * ZC * sizeof(float));       // concat features
    float* hbuf  = (float*)carve((size_t)N * WMAX * sizeof(float));     // per-layer x@W
    float* als   = (float*)carve((size_t)N * HMAX * sizeof(float));
    float* ald   = (float*)carve((size_t)N * HMAX * sizeof(float));
    float* mmax  = (float*)carve((size_t)N * HMAX * sizeof(float));
    float* denom = (float*)carve((size_t)N * HMAX * sizeof(float));
    float* exbuf = (float*)carve((size_t)EDGES * HMAX * sizeof(float));
    float* z1    = (float*)carve((size_t)N * 384 * sizeof(float));
    float* sums  = (float*)carve((size_t)G * 384 * sizeof(float));
    float* cntb  = (float*)carve((size_t)G * sizeof(float));
    float* zp    = (float*)carve((size_t)G * 384 * sizeof(float));
    float* z2    = (float*)carve((size_t)G * 256 * sizeof(float));

    // ---- GAT layers ----
    int aoff = 0, Kc = 3, coff = 0;
    for (int L = 0; L < 6; ++L) {
        const int H = HH[L], F = CO[L], width = H * F;

        // h = inA @ W
        if (L == 0) {
            lin_k3<<<gsz((long long)N * width, 256), 256, 0, stream>>>(x, Wl[0], hbuf, N, width);
        } else {
            launch_gemm(width, z, Wl[L], hbuf, N, Kc, aoff, stream);
        }

        const int nh = N * H;
        attn_logits<<<gsz(nh, 256), 256, 0, stream>>>(hbuf, ASl[L], ADl[L], als, ald, N, H, F);
        init_attn_state<<<gsz(nh, 256), 256, 0, stream>>>(mmax, denom, nh);

        init_out_bias<<<gsz((long long)N * width, 256), 256, 0, stream>>>(
            z, Bl[L], N, width, ZC, coff);

        const long long eh = (long long)EDGES * H;
        edge_max<<<gsz(eh, 256), 256, 0, stream>>>(src, dst, E, EDGES, als, ald, mmax, H);
        edge_exp<<<gsz(eh, 256), 256, 0, stream>>>(src, dst, E, EDGES, als, ald, mmax,
                                                   exbuf, denom, H);
        edge_aggr<<<gsz((long long)EDGES * width, 256), 256, 0, stream>>>(
            src, dst, E, EDGES, hbuf, exbuf, denom, z, H, F, ZC, coff);

        // next layer reads this layer's output block of z
        aoff = coff; Kc = width;
        coff += width;
    }

    // ---- head: fc1 (WMMA) + bias/BN/ReLU ----
    launch_gemm(384, z, fc1_W, z1, N, ZC, 0, stream);
    bias_bn_relu<<<gsz((long long)N * 384, 256), 256, 0, stream>>>(
        z1, fc1_b, bn_g, bn_b, bn_m, bn_v, N, 384);

    // ---- global mean pool ----
    pool_zero<<<gsz(G * 384, 256), 256, 0, stream>>>(sums, cntb, G, 384);
    pool_cnt<<<gsz(N, 256), 256, 0, stream>>>(batch, cntb, N);
    pool_sum<<<gsz((long long)N * 384, 256), 256, 0, stream>>>(batch, z1, sums, N, 384);
    pool_div<<<gsz(G * 384, 256), 256, 0, stream>>>(sums, cntb, zp, G, 384);

    // ---- fc2 + lin ----
    fc2_relu<<<gsz(G * 256, 256), 256, 0, stream>>>(zp, fc2_W, fc2_b, z2, G, 384, 256);
    lin_sigmoid<<<gsz(G, 64), 64, 0, stream>>>(z2, lin_W, lin_b, (float*)d_out, G, 256);
}
